// MaskedMultiHeadAttention_39977555591545
// MI455X (gfx1250) — compile-verified, matrix operand order checked by reading, not run on hardware
//
#include <hip/hip_runtime.h>
#include <hip/hip_bf16.h>

// ---------------------------------------------------------------------------
// Masked multi-head attention (causal), CDNA5 / gfx1250, wave32 WMMA path.
//   B=4, S=2048, hidden=1024, H=16 heads, D=64 head dim.
// All matmuls run on v_wmma_f32_16x16x32_f16 (f16 inputs, fp32 accumulate).
// GEMMs are LDS-blocked (128x128 per 8-wave workgroup) with the weight panel
// staged by double-buffered 2D Tensor-Data-Mover descriptors; attention is
// flash-style with double-buffered TDM K/V tiles shared by 4 waves.
// ---------------------------------------------------------------------------

typedef _Float16 v16h __attribute__((ext_vector_type(16)));
typedef float    v8f  __attribute__((ext_vector_type(8)));
typedef _Float16 h2   __attribute__((ext_vector_type(2)));
typedef unsigned int u32x4 __attribute__((ext_vector_type(4)));
typedef int          i32x4 __attribute__((ext_vector_type(4)));
typedef int          i32x8 __attribute__((ext_vector_type(8)));

#define NB 4
#define NS 2048
#define NHID 1024
#define NH 16
#define ND 64
#define BH (NB * NH)          // 64
#define ROWS (NB * NS)        // 8192 token rows

// ---- WMMA fragment helpers (ISA 7.12.2 layouts, wave32) --------------------
// 16-bit A fragment (16xK): lane l holds row m=l&15; VGPR v holds
//   k = (v>>2)*16 + (l>>4)*8 + (v&3)*2  (+kbase), pair {k,k+1}.
// B fragment (Kx16) uses the identical pattern with lane l&15 = column n,
// i.e. it loads Bt (NxK row-major) exactly like an A fragment.
static __device__ __forceinline__ v16h load_frag_row(const _Float16* rowPtr,
                                                     int kbase, int lane) {
  v16h f;
  const int kh = (lane >> 4) << 3;
#pragma unroll
  for (int v = 0; v < 8; ++v) {
    const int k = kbase + ((v >> 2) << 4) + kh + ((v & 3) << 1);
    h2 p = *(const h2*)(rowPtr + k);
    f[2 * v]     = p[0];
    f[2 * v + 1] = p[1];
  }
  return f;
}

// B fragment where the K dimension is strided in memory (e.g. V tile in LDS:
// element (k, col) at base[k*stride + col]).
static __device__ __forceinline__ v16h load_frag_kstride(const _Float16* base,
                                                         int col, int stride,
                                                         int lane) {
  v16h f;
  const int kh = (lane >> 4) << 3;
#pragma unroll
  for (int v = 0; v < 8; ++v) {
    const int k = ((v >> 2) << 4) + kh + ((v & 3) << 1);
    f[2 * v]     = base[k * stride + col];
    f[2 * v + 1] = base[(k + 1) * stride + col];
  }
  return f;
}

static __device__ __forceinline__ v8f wmma_f16(v16h a, v16h b, v8f c) {
  return __builtin_amdgcn_wmma_f32_16x16x32_f16(false, a, false, b,
                                                (short)0, c, false, false);
}

// ---- Tensor Data Mover: 2D tile (tile_d1 rows of tile_d0 elems) -> LDS -----
#if defined(__has_builtin)
#if __has_builtin(__builtin_amdgcn_tensor_load_to_lds) && \
    __has_builtin(__builtin_amdgcn_s_wait_tensorcnt)
#define USE_TDM 1
#endif
#endif

#ifdef USE_TDM
// dsize_code: 0=1B,1=2B,2=4B,3=8B. LDS receives packed rows (no padding).
static __device__ __forceinline__ void tdm_load_2d(
    unsigned lds_off, const void* gptr, unsigned dsize_code,
    unsigned tile_d0, unsigned tile_d1,
    unsigned long long tensor_d0, unsigned long long tensor_d1,
    unsigned long long stride0) {
  const unsigned long long ga = (unsigned long long)gptr;
  // D# group 0: count=1 (valid user descriptor), lds_addr, global_addr, type=2
  u32x4 g0;
  g0[0] = 1u;
  g0[1] = lds_off;
  g0[2] = (unsigned)(ga & 0xFFFFFFFFu);
  g0[3] = (unsigned)((ga >> 32) & 0x01FFFFFFu) | (2u << 30);
  // D# group 1
  i32x8 g1;
  g1[0] = (int)(dsize_code << 16);                         // wg_mask=0
  g1[1] = (int)((tensor_d0 & 0xFFFFu) << 16);              // dim0 lo16
  g1[2] = (int)(((tensor_d0 >> 16) & 0xFFFFu) |
                ((tensor_d1 & 0xFFFFu) << 16));            // dim0 hi / dim1 lo
  g1[3] = (int)(((tensor_d1 >> 16) & 0xFFFFu) |
                ((tile_d0 & 0xFFFFu) << 16));              // dim1 hi / tile0
  g1[4] = (int)(tile_d1 & 0xFFFFu);                        // tile1, tile2=0
  g1[5] = (int)(stride0 & 0xFFFFFFFFu);                    // dim0 stride lo32
  g1[6] = (int)((stride0 >> 32) & 0xFFFFu);                // stride hi / d1s lo
  g1[7] = 0;
  i32x4 z4 = {0, 0, 0, 0};
#if __clang_major__ >= 23
  i32x8 z8 = {0, 0, 0, 0, 0, 0, 0, 0};
  __builtin_amdgcn_tensor_load_to_lds(g0, g1, z4, z4, z8, 0);
#else
  __builtin_amdgcn_tensor_load_to_lds(g0, g1, z4, z4, 0);
#endif
}
#endif

static __device__ __forceinline__ unsigned lds_off_of(const void* p) {
  return (unsigned)(unsigned long long)p;
}

// ---- Kernel 1: x fp32 -> f16 ----------------------------------------------
__global__ void k_cvt_x(const float* __restrict__ x, _Float16* __restrict__ xh) {
  const int i = blockIdx.x * 256 + threadIdx.x;
  xh[i] = (_Float16)x[i];
}

// ---- Kernel 2: transpose + convert weights --------------------------------
__global__ void k_prep_w(const float* __restrict__ Wq, const float* __restrict__ Wk,
                         const float* __restrict__ Wv, const float* __restrict__ Wo,
                         _Float16* __restrict__ wqkvt, _Float16* __restrict__ wot) {
  const int i = blockIdx.x * 256 + threadIdx.x;     // [0, 4*1024*1024)
  const int mi  = i >> 20;
  const int rem = i & 1048575;
  const int k = rem >> 10;          // slow index; n is fast -> coalesced source reads
  const int n = rem & 1023;
  const float* src = (mi == 0) ? Wq : (mi == 1) ? Wk : (mi == 2) ? Wv : Wo;
  const float val = src[k * 1024 + n];
  if (mi < 3) wqkvt[((size_t)mi << 20) + (size_t)n * 1024 + k] = (_Float16)val;
  else        wot[(size_t)n * 1024 + k] = (_Float16)val;
}

// ---- Kernel 3: fused QKV projection (LDS-blocked, TDM double-buffered) -----
// 8 waves, 128(M) x 128(N) tile; W k-slice (128x32, strided) staged per step.
__global__ void __launch_bounds__(256)
k_qkv_gemm(const _Float16* __restrict__ xh,
           const _Float16* __restrict__ wt,
           _Float16* __restrict__ qkvh) {
  __shared__ _Float16 wTile[2][128 * 32];   // 2 x 8 KB

  const int mg = blockIdx.x / 24;           // 64 groups of 128 rows
  const int ct = blockIdx.x % 24;           // 24 groups of 128 cols (3072)
  const int wid  = threadIdx.x >> 5;
  const int lane = threadIdx.x & 31;
  const int m0 = mg * 128 + wid * 16;
  const int n0 = ct * 128;

  v8f acc[8] = {};
  const _Float16* arow = xh + (size_t)(m0 + (lane & 15)) * NHID;

#ifdef USE_TDM
  if (wid == 0)
    tdm_load_2d(lds_off_of(&wTile[0][0]), wt + (size_t)n0 * NHID,
                1, 32, 128, NHID, 3072, NHID);
#endif
  for (int i = 0; i < 32; ++i) {
    const int cur = i & 1;
#ifdef USE_TDM
    if (wid == 0) {
      if (i + 1 < 32) {
        tdm_load_2d(lds_off_of(&wTile[cur ^ 1][0]),
                    wt + (size_t)n0 * NHID + (i + 1) * 32,
                    1, 32, 128, NHID, 3072, NHID);
        __builtin_amdgcn_s_wait_tensorcnt(1);   // oldest (cur) complete
      } else {
        __builtin_amdgcn_s_wait_tensorcnt(0);
      }
    }
#else
    {
#pragma unroll
      for (int j = 0; j < 2; ++j) {
        const int idx = j * 256 + threadIdx.x;   // 512 x 16B = 8 KB
        const int row = idx >> 2, ch = idx & 3;
        *(u32x4*)&wTile[cur][row * 32 + ch * 8] =
            *(const u32x4*)&wt[(size_t)(n0 + row) * NHID + i * 32 + ch * 8];
      }
    }
#endif
    __syncthreads();
    const v16h a = load_frag_row(arow, i * 32, lane);
    const _Float16* wb = wTile[cur];
#pragma unroll
    for (int t = 0; t < 8; ++t) {
      const v16h b = load_frag_row(wb + (t * 16 + (lane & 15)) * 32, 0, lane);
      acc[t] = wmma_f16(a, b, acc[t]);
    }
    __syncthreads();
  }

  const int mloc = (lane >> 4) << 3;
  const int ncol = lane & 15;
#pragma unroll
  for (int t = 0; t < 8; ++t) {
#pragma unroll
    for (int r = 0; r < 8; ++r) {
      const int m = m0 + mloc + r;            // global token row
      const int n = n0 + t * 16 + ncol;       // [0, 3072)
      const int sel = n >> 10;                // 0=q,1=k,2=v
      const int c = n & 1023;
      const int h = c >> 6, d = c & 63;
      const int b = m >> 11, s = m & 2047;
      qkvh[((((size_t)sel * NB + b) * NH + h) * NS + s) * ND + d] =
          (_Float16)acc[t][r];
    }
  }
}

// ---- Kernel 4: causal flash attention -------------------------------------
// 128 threads = 4 waves per block; block covers 64 queries of one (b,h).
// K/V 32-key tiles staged per block via double-buffered TDM loads.
__global__ void __launch_bounds__(128)
k_attention(const _Float16* __restrict__ qkvh, _Float16* __restrict__ ctxh) {
  __shared__ _Float16 kT[2][32 * ND];     // 2 x 4 KB
  __shared__ _Float16 vT[2][32 * ND];     // 2 x 4 KB
  __shared__ _Float16 pT[4 * 16 * 32];    // 4 KB: per-wave P bounce

  const int bh = blockIdx.x >> 5;         // b*H + h  in [0,64)
  const int qg = blockIdx.x & 31;         // 64-query group in [0,32)
  const int wid  = threadIdx.x >> 5;
  const int lane = threadIdx.x & 31;
  const int q0_block = qg * 64;
  const int m0 = q0_block + wid * 16;     // this wave's first query row

  const _Float16* qptr = qkvh + (size_t)(0 * BH + bh) * NS * ND;
  const _Float16* kptr = qkvh + (size_t)(1 * BH + bh) * NS * ND;
  const _Float16* vptr = qkvh + (size_t)(2 * BH + bh) * NS * ND;
  _Float16* pW = pT + wid * (16 * 32);

  // Preload the Q fragments once (2 k-steps covering D=64).
  v16h qf[2];
#pragma unroll
  for (int ds2 = 0; ds2 < 2; ++ds2)
    qf[ds2] = load_frag_row(qptr + (size_t)(m0 + (lane & 15)) * ND, ds2 * 32, lane);

  v8f accO[4] = {};
  float mrow[8], lrow[8];
#pragma unroll
  for (int r = 0; r < 8; ++r) { mrow[r] = -__builtin_inff(); lrow[r] = 0.f; }

  const int nkb = (q0_block + 95) >> 5;   // key blocks covering keys <= q0+63
  const int qrow_base = m0 + ((lane >> 4) << 3);
  const int nkey = lane & 15;

#ifdef USE_TDM
  if (wid == 0) {
    tdm_load_2d(lds_off_of(&kT[0][0]), kptr, 1, 2048, 1, 2048, 1, 2048);
    tdm_load_2d(lds_off_of(&vT[0][0]), vptr, 1, 2048, 1, 2048, 1, 2048);
  }
#endif
  for (int kb = 0; kb < nkb; ++kb) {
    const int kbase = kb * 32;
    const int cur = kb & 1;
#ifdef USE_TDM
    if (wid == 0) {
      if (kb + 1 < nkb) {
        tdm_load_2d(lds_off_of(&kT[cur ^ 1][0]),
                    kptr + (size_t)(kbase + 32) * ND, 1, 2048, 1, 2048, 1, 2048);
        tdm_load_2d(lds_off_of(&vT[cur ^ 1][0]),
                    vptr + (size_t)(kbase + 32) * ND, 1, 2048, 1, 2048, 1, 2048);
        __builtin_amdgcn_s_wait_tensorcnt(2);  // cur's two loads complete
      } else {
        __builtin_amdgcn_s_wait_tensorcnt(0);
      }
    }
#else
    {
      const u32x4* ks = (const u32x4*)(kptr + (size_t)kbase * ND);
      const u32x4* vs = (const u32x4*)(vptr + (size_t)kbase * ND);
      u32x4* kd = (u32x4*)kT[cur];
      u32x4* vd = (u32x4*)vT[cur];
#pragma unroll
      for (int i = 0; i < 2; ++i) {
        kd[i * 128 + threadIdx.x] = ks[i * 128 + threadIdx.x];
        vd[i * 128 + threadIdx.x] = vs[i * 128 + threadIdx.x];
      }
    }
#endif
    __syncthreads();

    // ---- scores S = Q K^T over this 32-key block (4 WMMA, K from LDS) ----
    v8f s0 = {}, s1 = {};
#pragma unroll
    for (int ds2 = 0; ds2 < 2; ++ds2) {
      const v16h bk0 = load_frag_row(kT[cur] + (size_t)nkey * ND, ds2 * 32, lane);
      const v16h bk1 = load_frag_row(kT[cur] + (size_t)(16 + nkey) * ND, ds2 * 32, lane);
      s0 = wmma_f16(qf[ds2], bk0, s0);
      s1 = wmma_f16(qf[ds2], bk1, s1);
    }

    // ---- scale + causal mask + online softmax ----
    // C-layout: row m = (lane>>4)*8 + r lives in one 16-lane half -> reduce
    // with shfl_xor strides 1,2,4,8. Fully-masked blocks degenerate to
    // alpha=1, p=0 (harmless).
#pragma unroll
    for (int r = 0; r < 8; ++r) {
      const int q = qrow_base + r;
      float a0 = (float)s0[r] * 0.125f;
      float a1 = (float)s1[r] * 0.125f;
      if (kbase + nkey > q)      a0 = -__builtin_inff();
      if (kbase + 16 + nkey > q) a1 = -__builtin_inff();

      float mx = fmaxf(a0, a1);
#pragma unroll
      for (int off = 1; off < 16; off <<= 1)
        mx = fmaxf(mx, __shfl_xor(mx, off, 32));
      const float mn = fmaxf(mrow[r], mx);

      const float p0 = __expf(a0 - mn);
      const float p1 = __expf(a1 - mn);
      float sum = p0 + p1;
#pragma unroll
      for (int off = 1; off < 16; off <<= 1)
        sum += __shfl_xor(sum, off, 32);

      const float alpha = __expf(mrow[r] - mn);
      lrow[r] = lrow[r] * alpha + sum;
      mrow[r] = mn;
#pragma unroll
      for (int t = 0; t < 4; ++t) accO[t][r] = accO[t][r] * alpha;

      const int prow = ((lane >> 4) << 3) + r;
      pW[prow * 32 + nkey]      = (_Float16)p0;
      pW[prow * 32 + 16 + nkey] = (_Float16)p1;
    }
    // pW is private to this wave; LDS ops of one wave are in-order, so the
    // A-fragment read below sees the stores without a block barrier.

    // ---- O += P V  (4 WMMA, one per 16-wide slice of D) ----
    const v16h pa = load_frag_row(pW + (lane & 15) * 32, 0, lane);
#pragma unroll
    for (int t = 0; t < 4; ++t) {
      const v16h bv = load_frag_kstride(vT[cur], t * 16 + (lane & 15), ND, lane);
      accO[t] = wmma_f16(pa, bv, accO[t]);
    }
    __syncthreads();   // all waves done with cur tiles before they refill
  }

  // ---- finalize: O /= l, scatter to ctx[B,S,H*D] ----
  const int b = bh >> 4, h = bh & 15;
#pragma unroll
  for (int r = 0; r < 8; ++r) {
    const float inv = 1.f / lrow[r];
    const int s = qrow_base + r;
#pragma unroll
    for (int t = 0; t < 4; ++t) {
      const int d = t * 16 + (lane & 15);
      ctxh[((size_t)b * NS + s) * NHID + h * ND + d] =
          (_Float16)(accO[t][r] * inv);
    }
  }
}

// ---- Kernel 5: output projection + bias (LDS-blocked, TDM, fp32 out) ------
__global__ void __launch_bounds__(256)
k_out_gemm(const _Float16* __restrict__ ctxh,
           const _Float16* __restrict__ wot,
           const float* __restrict__ bo,
           float* __restrict__ out) {
  __shared__ _Float16 wTile[2][128 * 32];   // 2 x 8 KB

  const int mg = blockIdx.x >> 3;           // 64 groups of 128 rows
  const int ct = blockIdx.x & 7;            // 8 groups of 128 cols (1024)
  const int wid  = threadIdx.x >> 5;
  const int lane = threadIdx.x & 31;
  const int m0 = mg * 128 + wid * 16;
  const int n0 = ct * 128;

  v8f acc[8] = {};
  const _Float16* arow = ctxh + (size_t)(m0 + (lane & 15)) * NHID;

#ifdef USE_TDM
  if (wid == 0)
    tdm_load_2d(lds_off_of(&wTile[0][0]), wot + (size_t)n0 * NHID,
                1, 32, 128, NHID, 1024, NHID);
#endif
  for (int i = 0; i < 32; ++i) {
    const int cur = i & 1;
#ifdef USE_TDM
    if (wid == 0) {
      if (i + 1 < 32) {
        tdm_load_2d(lds_off_of(&wTile[cur ^ 1][0]),
                    wot + (size_t)n0 * NHID + (i + 1) * 32,
                    1, 32, 128, NHID, 1024, NHID);
        __builtin_amdgcn_s_wait_tensorcnt(1);
      } else {
        __builtin_amdgcn_s_wait_tensorcnt(0);
      }
    }
#else
    {
#pragma unroll
      for (int j = 0; j < 2; ++j) {
        const int idx = j * 256 + threadIdx.x;
        const int row = idx >> 2, ch = idx & 3;
        *(u32x4*)&wTile[cur][row * 32 + ch * 8] =
            *(const u32x4*)&wot[(size_t)(n0 + row) * NHID + i * 32 + ch * 8];
      }
    }
#endif
    __syncthreads();
    const v16h a = load_frag_row(arow, i * 32, lane);
    const _Float16* wb = wTile[cur];
#pragma unroll
    for (int t = 0; t < 8; ++t) {
      const v16h b = load_frag_row(wb + (t * 16 + (lane & 15)) * 32, 0, lane);
      acc[t] = wmma_f16(a, b, acc[t]);
    }
    __syncthreads();
  }

  const int mloc = (lane >> 4) << 3;
  const int ncol = lane & 15;
#pragma unroll
  for (int t = 0; t < 8; ++t) {
    const int n = n0 + t * 16 + ncol;
    const float bias = bo[n];
#pragma unroll
    for (int r = 0; r < 8; ++r) {
      const int m = m0 + mloc + r;
      out[(size_t)m * NHID + n] = acc[t][r] + bias;
    }
  }
}

// ---------------------------------------------------------------------------
extern "C" void kernel_launch(void* const* d_in, const int* in_sizes, int n_in,
                              void* d_out, int out_size, void* d_ws, size_t ws_size,
                              hipStream_t stream) {
  const float* x  = (const float*)d_in[0];
  const float* Wq = (const float*)d_in[1];
  const float* Wk = (const float*)d_in[2];
  const float* Wv = (const float*)d_in[3];
  const float* Wo = (const float*)d_in[4];
  const float* bo = (const float*)d_in[5];
  float* out = (float*)d_out;

  // workspace carve-out (bytes):
  //   xh: 16 MiB | wqkvt: 6 MiB | wot: 2 MiB | qkvh: 48 MiB [3][B][H][S][D]
  //   ctxh: 16 MiB   (total ~88 MiB)
  char* ws = (char*)d_ws;
  _Float16* xh    = (_Float16*)(ws);
  _Float16* wqkvt = (_Float16*)(ws + (size_t)16777216);
  _Float16* wot   = (_Float16*)(ws + (size_t)16777216 + 6291456);
  _Float16* qkvh  = (_Float16*)(ws + (size_t)16777216 + 6291456 + 2097152);
  _Float16* ctxh  = (_Float16*)(ws + (size_t)16777216 + 6291456 + 2097152 + 50331648);

  k_cvt_x    <<<ROWS * NHID / 256, 256, 0, stream>>>(x, xh);
  k_prep_w   <<<4 * 1024 * 1024 / 256, 256, 0, stream>>>(Wq, Wk, Wv, Wo, wqkvt, wot);
  k_qkv_gemm <<<64 * 24, 256, 0, stream>>>(xh, wqkvt, qkvh);
  k_attention<<<BH * (NS / 64), 128, 0, stream>>>(qkvh, ctxh);
  k_out_gemm <<<64 * 8, 256, 0, stream>>>(ctxh, wot, bo, out);
}
